// ContrastiveLoss_67095979099189
// MI455X (gfx1250) — matrix operand order repaired and checked by reading, not compiled, not run on hardware
//
#include <hip/hip_runtime.h>
#include <hip/hip_bf16.h>

#define B_ROWS 8192
#define DIM    512
#define EPS    1e-8f

#define JCHUNK 1024          // columns swept per block
#define NTILES (JCHUNK / 16) // 64 j-tiles per block

typedef __attribute__((ext_vector_type(16))) _Float16 v16h;
typedef __attribute__((ext_vector_type(8)))  _Float16 v8h;
typedef __attribute__((ext_vector_type(8)))  float    v8f;

typedef __attribute__((address_space(3))) const _Float16 lds_cf16;

__device__ __forceinline__ float wave_reduce_add32(float v) {
    v += __shfl_xor(v, 1);
    v += __shfl_xor(v, 2);
    v += __shfl_xor(v, 4);
    v += __shfl_xor(v, 8);
    v += __shfl_xor(v, 16);
    return v;
}

__device__ __forceinline__ unsigned lds_offset(const _Float16* p) {
    return (unsigned)(uintptr_t)(lds_cf16*)p;   // addrspacecast -> 32-bit LDS offset
}

// ---------------------------------------------------------------------------
// Kernel 1: per-row norms, rowwise cosine(x1,x2), normalized f16 rows.
// One wave per row; 8 waves per 256-thread block.
// ---------------------------------------------------------------------------
__global__ void __launch_bounds__(256)
prep_kernel(const float* __restrict__ x1, const float* __restrict__ x2,
            _Float16* __restrict__ y, float* __restrict__ sim12,
            float* __restrict__ simneg, float* __restrict__ out) {
    const int wave = threadIdx.x >> 5;
    const int lane = threadIdx.x & 31;
    const int row  = blockIdx.x * 8 + wave;
    if (row >= B_ROWS) return;

    const float* __restrict__ r1 = x1 + (size_t)row * DIM;
    const float* __restrict__ r2 = x2 + (size_t)row * DIM;

    float ss1 = 0.f, ss2 = 0.f, d12 = 0.f;
    float v1[DIM / 32];
#pragma unroll
    for (int t = 0; t < DIM / 32; ++t) {
        const float a = r1[lane + 32 * t];
        const float b = r2[lane + 32 * t];
        v1[t] = a;
        ss1 = fmaf(a, a, ss1);
        ss2 = fmaf(b, b, ss2);
        d12 = fmaf(a, b, d12);
    }
    ss1 = wave_reduce_add32(ss1);
    ss2 = wave_reduce_add32(ss2);
    d12 = wave_reduce_add32(d12);

    const float n1  = sqrtf(ss1);
    const float n2  = sqrtf(ss2);
    const float rn1 = 1.0f / n1;   // norms ~ sqrt(512), never near eps here

#pragma unroll
    for (int t = 0; t < DIM / 32; ++t)
        y[(size_t)row * DIM + lane + 32 * t] = (_Float16)(v1[t] * rn1);

    if (lane == 0) {
        sim12[row]  = d12 / fmaxf(n1 * n2, EPS);
        simneg[row] = 0.f;
        if (row == 0) out[0] = 0.f;
    }
}

// ---------------------------------------------------------------------------
// Kernel 2: fused Gram + exp + row-sum, B-tile shared block-wide via LDS.
//   block -> 128 rows (wave w owns rows row0+16w, A register-resident:
//            128 VGPRs/wave) x one 1024-column j-chunk in 16-column tiles.
//   B tile (16 x 512 f16 = 16 KB) double-buffered in LDS, staged with
//   global_load_async_to_lds_b128.
//   Single-barrier pipeline per phase:
//       wait(cur tile loads) -> barrier -> issue stage(next) -> compute(cur)
//   Barrier at phase t also proves all waves finished computing from the
//   other buffer in phase t-1, making the post-barrier overwrite WAR-safe.
// ---------------------------------------------------------------------------
__global__ void __launch_bounds__(256)
gram_expsum_kernel(const _Float16* __restrict__ y, float* __restrict__ simneg) {
    __shared__ __align__(16) _Float16 tileBuf[2][16 * DIM];  // 2 x 16 KB

    const int tid   = threadIdx.x;
    const int wave  = tid >> 5;
    const int lane  = tid & 31;
    const int laneN = lane & 15;
    const int hi    = lane >> 4;

    const int mblock = blockIdx.x >> 3;       // 64 M-blocks of 128 rows
    const int jchunk = blockIdx.x & 7;        // 8 j-chunks of 1024 cols
    const int row0   = mblock * 128 + wave * 16;
    const int j0base = jchunk * JCHUNK;

    // --- explicit A hoist: rows row0..row0+15, all K, 128 VGPRs ------------
    const _Float16* __restrict__ Arow = y + (size_t)(row0 + laneN) * DIM;
    v16h afrag[16];
#pragma unroll
    for (int ks = 0; ks < 16; ++ks) {
        union { v16h v; v8h h[2]; } a;
        const int offA = ks * 32 + hi * 16;
        a.h[0] = *(const v8h*)(Arow + offA);
        a.h[1] = *(const v8h*)(Arow + offA + 8);
        afrag[ks] = a.v;
    }

    // --- async stage of one 16-row B tile into LDS (4 x b128 per thread) ---
    auto stage = [&](int buf, int jt) {
        const unsigned base = lds_offset(&tileBuf[buf][0]);
#pragma unroll
        for (int i = 0; i < 4; ++i) {
            const int c  = tid + 256 * i;        // 16-byte chunk id, 0..1023
            const int r  = c >> 6;               // row within tile (64 chunks/row)
            const int cb = (c & 63) * 16;        // byte offset within row
            const unsigned long long ga =
                (unsigned long long)(const char*)(y + (size_t)(j0base + jt * 16 + r) * DIM) + cb;
            const unsigned la = base + (unsigned)c * 16;
            asm volatile("global_load_async_to_lds_b128 %0, %1, off"
                         :: "v"(la), "v"(ga) : "memory");
        }
    };

    float racc[8];
#pragma unroll
    for (int r = 0; r < 8; ++r) racc[r] = 0.f;

    stage(0, 0);
    for (int jt = 0; jt < NTILES; ++jt) {
        const int cur = jt & 1;
        // current tile's async loads (issued last phase) complete...
        asm volatile("s_wait_asynccnt 0x0" ::: "memory");
        // ...and every wave has both waited and finished last phase's compute
        __syncthreads();
        // prefetch next tile into the buffer freed by phase jt-1
        if (jt + 1 < NTILES) stage(cur ^ 1, jt + 1);

        const _Float16* Brow = &tileBuf[cur][laneN * DIM];
        // two independent accumulator chains (dot is linear; sum before exp)
        v8f c0 = {}, c1 = {};
#pragma unroll
        for (int ks = 0; ks < 16; ks += 2) {
            const v16h b0 = *(const v16h*)(Brow + (ks + 0) * 32 + hi * 16);
            const v16h b1 = *(const v16h*)(Brow + (ks + 1) * 32 + hi * 16);
            c0 = __builtin_amdgcn_wmma_f32_16x16x32_f16(
                     false, afrag[ks + 0], false, b0, (short)0, c0, false, false);
            c1 = __builtin_amdgcn_wmma_f32_16x16x32_f16(
                     false, afrag[ks + 1], false, b1, (short)0, c1, false, false);
        }
#pragma unroll
        for (int r = 0; r < 8; ++r) racc[r] += __expf(c0[r] + c1[r]);
    }

    // reduce across the 16 N-lanes of each half-wave (xor masks < 16 stay in-half)
#pragma unroll
    for (int r = 0; r < 8; ++r) {
        float v = racc[r];
        v += __shfl_xor(v, 1);
        v += __shfl_xor(v, 2);
        v += __shfl_xor(v, 4);
        v += __shfl_xor(v, 8);
        racc[r] = v;
    }
    if (lane == 0) {
#pragma unroll
        for (int r = 0; r < 8; ++r) atomicAdd(&simneg[row0 + r], racc[r]);
    } else if (lane == 16) {
#pragma unroll
        for (int r = 0; r < 8; ++r) atomicAdd(&simneg[row0 + 8 + r], racc[r]);
    }
}

// ---------------------------------------------------------------------------
// Kernel 3: subtract excluded shift term, assemble per-row loss, mean-reduce.
// ---------------------------------------------------------------------------
__global__ void __launch_bounds__(256)
finish_kernel(const _Float16* __restrict__ y, const float* __restrict__ sim12,
              const float* __restrict__ simneg, float* __restrict__ out) {
    const int wave = threadIdx.x >> 5;
    const int lane = threadIdx.x & 31;
    const int row  = blockIdx.x * 8 + wave;
    if (row >= B_ROWS) return;

    const int prev = (row == 0) ? (B_ROWS - 1) : (row - 1);
    const _Float16* __restrict__ yi = y + (size_t)row  * DIM;
    const _Float16* __restrict__ yp = y + (size_t)prev * DIM;

    float dot = 0.f;
#pragma unroll
    for (int t = 0; t < DIM / 32; ++t)
        dot = fmaf((float)yi[lane + 32 * t], (float)yp[lane + 32 * t], dot);
    dot = wave_reduce_add32(dot);

    if (lane == 0) {
        const float excl = __expf(dot);
        const float neg  = simneg[row] - excl;
        const float s12  = sim12[row];
        const float pos  = __expf(s12);
        // -log(pos/(pos+neg)) = log(pos+neg) - s12
        const float loss = __logf(pos + neg) - s12;
        atomicAdd(out, loss * (1.0f / (float)B_ROWS));
    }
}

extern "C" void kernel_launch(void* const* d_in, const int* in_sizes, int n_in,
                              void* d_out, int out_size, void* d_ws, size_t ws_size,
                              hipStream_t stream) {
    const float* x1 = (const float*)d_in[0];
    const float* x2 = (const float*)d_in[1];
    float* out = (float*)d_out;

    _Float16* y   = (_Float16*)d_ws;                                   // 8 MB
    float* sim12  = (float*)((char*)d_ws + (size_t)B_ROWS * DIM * 2);  // 32 KB
    float* simneg = sim12 + B_ROWS;                                    // 32 KB

    prep_kernel<<<B_ROWS / 8, 256, 0, stream>>>(x1, x2, y, sim12, simneg, out);
    gram_expsum_kernel<<<(B_ROWS / 128) * 8, 256, 0, stream>>>(y, simneg);
    finish_kernel<<<B_ROWS / 8, 256, 0, stream>>>(y, sim12, simneg, out);
}
